// PairStack_63531156242734
// MI455X (gfx1250) — matrix operand order repaired and checked by reading, not compiled, not run on hardware
//
#include <hip/hip_runtime.h>

typedef unsigned short u16;
typedef unsigned int   u32;
typedef __attribute__((ext_vector_type(2)))  u32   u32x2;
typedef __attribute__((ext_vector_type(4)))  u32   u32x4;
typedef __attribute__((ext_vector_type(4)))  float f32x4;
typedef __attribute__((ext_vector_type(8)))  float v8f;
typedef __attribute__((ext_vector_type(16))) __bf16 v16bf;

union Frag { u32x4 u[2]; v16bf v; };

#define WMMA_BF16(A, B, C) \
  __builtin_amdgcn_wmma_f32_16x16x32_bf16(false, (A), false, (B), (short)0, (C), false, false)

__device__ __forceinline__ u16 f2bf(float f) {
  u32 u = __float_as_uint(f);
  u32 r = u + 0x7fffu + ((u >> 16) & 1u);
  return (u16)(r >> 16);
}
__device__ __forceinline__ float bf2f(u16 h) { return __uint_as_float(((u32)h) << 16); }

__device__ __forceinline__ float wsum32(float v) {
  for (int o = 16; o > 0; o >>= 1) v += __shfl_xor(v, o, 32);
  return v;
}
__device__ __forceinline__ float hmax16(float v) {
  for (int o = 1; o < 16; o <<= 1) v = fmaxf(v, __shfl_xor(v, o, 32));
  return v;
}
__device__ __forceinline__ float hsum16(float v) {
  for (int o = 1; o < 16; o <<= 1) v += __shfl_xor(v, o, 32);
  return v;
}
__device__ __forceinline__ int swapm(int m) { return ((m & 255) << 8) | (m >> 8); }

// ---------------- LayerNorm over last dim (128), f32 in -> bf16 out --------
// One wave per row. transposed: read source row index-swapped ((i,j)->(j,i)).
__global__ __launch_bounds__(256) void ln_k(const float* __restrict__ X,
                                            const float* __restrict__ g,
                                            const float* __restrict__ b,
                                            u16* __restrict__ Z, int transposed) {
  int row  = blockIdx.x * 8 + (threadIdx.x >> 5);
  int lane = threadIdx.x & 31;
  int rrow = transposed ? swapm(row) : row;
  f32x4 x  = *(const f32x4*)(X + (size_t)rrow * 128 + lane * 4);
  float mean = wsum32(x.x + x.y + x.z + x.w) * (1.f / 128.f);
  f32x4 d  = x - mean;
  float var = wsum32(d.x * d.x + d.y * d.y + d.z * d.z + d.w * d.w) * (1.f / 128.f);
  float rstd = rsqrtf(var + 1e-5f);
  f32x4 gg = *(const f32x4*)(g + lane * 4);
  f32x4 bb = *(const f32x4*)(b + lane * 4);
  f32x4 y  = d * rstd * gg + bb;
  u32x2 p;
  p.x = (u32)f2bf(y.x) | ((u32)f2bf(y.y) << 16);
  p.y = (u32)f2bf(y.z) | ((u32)f2bf(y.w) << 16);
  *(u32x2*)(Z + (size_t)row * 128 + lane * 4) = p;
}

// ---------------- Weight prep: f32 W[K][N] -> bf16 Wt[N][K] (K-contiguous) --
__global__ void prep_wt(const float* __restrict__ W, u16* __restrict__ Wt, int K, int N) {
  int idx = blockIdx.x * 256 + threadIdx.x;
  if (idx >= K * N) return;
  int n = idx / K, k = idx - n * K;
  Wt[idx] = f2bf(W[(size_t)k * N + n]);
}

// ---------------- tri_bias: tb[h][m] = dot(Z[m], Wb[h])  (H=4) --------------
__global__ __launch_bounds__(256) void tb_k(const u16* __restrict__ Z,
                                            const float* __restrict__ Wb,
                                            float* __restrict__ tb) {
  int row  = blockIdx.x * 8 + (threadIdx.x >> 5);
  int lane = threadIdx.x & 31;
  u32x2 zz = *(const u32x2*)(Z + (size_t)row * 128 + lane * 4);
  float z0 = bf2f(zz.x & 0xffff), z1 = bf2f(zz.x >> 16);
  float z2 = bf2f(zz.y & 0xffff), z3 = bf2f(zz.y >> 16);
  for (int h = 0; h < 4; h++) {
    f32x4 w = *(const f32x4*)(Wb + h * 128 + lane * 4);
    float s = wsum32(z0 * w.x + z1 * w.y + z2 * w.z + z3 * w.w);
    if (lane == 0) tb[h * 65536 + row] = s;
  }
}

// ---------------- elementwise gate*o -> bf16 --------------------------------
__global__ __launch_bounds__(256) void gatemul_k(const float* __restrict__ g,
                                                 const float* __restrict__ o,
                                                 u16* __restrict__ out) {
  int idx = (blockIdx.x * 256 + threadIdx.x) * 4;
  f32x4 gv = *(const f32x4*)(g + idx);
  f32x4 ov = *(const f32x4*)(o + idx);
  f32x4 r  = gv * ov;
  u32x2 p;
  p.x = (u32)f2bf(r.x) | ((u32)f2bf(r.y) << 16);
  p.y = (u32)f2bf(r.z) | ((u32)f2bf(r.w) << 16);
  *(u32x2*)(out + idx) = p;
}

// ---------------- Generic WMMA GEMM: [65536,K]bf16 x Wt[N][K]bf16 -----------
// Block = 8 waves; each wave computes a 16x128 output strip; K loop step 32.
// EPI: 0 sigmoid->f32 row   1 (v+b)*gate*mask -> bf16 channel-major [c][m(T)]
//      2 relu -> bf16 row   3 out = pairin + v (+bias), optional index swap
//      4 out = pairin + gate*(v+bias)          5 bf16 row (no bias -> pass 0)
//      6 bf16 V-layout [i][h][c][k]
template <int EPI>
__global__ __launch_bounds__(256) void gemm_k(const u16* __restrict__ A, int K,
                                              const u16* __restrict__ Wt,
                                              const float* __restrict__ bias,
                                              const float* __restrict__ mask,
                                              const float* __restrict__ gate,
                                              const float* __restrict__ pairin,
                                              float* __restrict__ outf,
                                              u16* __restrict__ outb,
                                              int Nout, int tflag) {
  int wv = threadIdx.x >> 5, lane = threadIdx.x & 31;
  int hf = lane >> 4, l15 = lane & 15;
  int m0 = blockIdx.x * 128 + wv * 16;
  int n0 = blockIdx.y * 128;
  int arow = m0 + l15;

  v8f acc[8];
#pragma unroll
  for (int t = 0; t < 8; t++) acc[t] = (v8f){0, 0, 0, 0, 0, 0, 0, 0};

  const u16* ab = A + (size_t)arow * K + hf * 16;
  for (int kk = 0; kk < K; kk += 32) {
    Frag af;
    af.u[0] = *(const u32x4*)(ab + kk);
    af.u[1] = *(const u32x4*)(ab + kk + 8);
#pragma unroll
    for (int t = 0; t < 8; t++) {
      Frag bf;
      const u16* bp = Wt + (size_t)(n0 + t * 16 + l15) * K + kk + hf * 16;
      bf.u[0] = *(const u32x4*)(bp);
      bf.u[1] = *(const u32x4*)(bp + 8);
      acc[t] = WMMA_BF16(af.v, bf.v, acc[t]);
    }
  }

#pragma unroll
  for (int t = 0; t < 8; t++) {
    int n = n0 + t * 16 + l15;
    float bv = bias ? bias[n] : 0.f;
#pragma unroll
    for (int r = 0; r < 8; r++) {
      int m = m0 + r + hf * 8;
      float v = acc[t][r] + bv;
      if constexpr (EPI == 0) {
        outf[(size_t)m * Nout + n] = 1.f / (1.f + __expf(-v));
      } else if constexpr (EPI == 1) {
        int mm = tflag ? swapm(m) : m;
        outb[(size_t)n * 65536 + mm] = f2bf(v * gate[(size_t)m * 128 + n] * mask[m]);
      } else if constexpr (EPI == 2) {
        outb[(size_t)m * Nout + n] = f2bf(fmaxf(v, 0.f));
      } else if constexpr (EPI == 3) {
        int mm = tflag ? swapm(m) : m;
        size_t ix = (size_t)mm * 128 + n;
        outf[ix] = pairin[ix] + v;
      } else if constexpr (EPI == 4) {
        size_t ix = (size_t)m * 128 + n;
        outf[ix] = pairin[ix] + gate[ix] * v;
      } else if constexpr (EPI == 5) {
        outb[(size_t)m * 128 + n] = f2bf(v);
      } else if constexpr (EPI == 6) {
        outb[(size_t)((m >> 8) * 128 + n) * 256 + (m & 255)] = f2bf(v);
      }
    }
  }
}

// ---------------- Triangle einsum: per channel c, pm_c = L_c * R_c^T --------
// L[c][256][256] row-major (K-contig), R[c][256][256] already B^T (K-contig).
__global__ __launch_bounds__(256) void einsum_k(const u16* __restrict__ L,
                                                const u16* __restrict__ R,
                                                float* __restrict__ pm) {
  int wv = threadIdx.x >> 5, lane = threadIdx.x & 31;
  int hf = lane >> 4, l15 = lane & 15;
  int c  = blockIdx.z;
  int i0 = blockIdx.y * 128 + wv * 16;
  int j0 = blockIdx.x * 128;
  const u16* Ab = L + (size_t)c * 65536;
  const u16* Bb = R + (size_t)c * 65536;

  v8f acc[8];
#pragma unroll
  for (int t = 0; t < 8; t++) acc[t] = (v8f){0, 0, 0, 0, 0, 0, 0, 0};

  const u16* ap = Ab + (size_t)(i0 + l15) * 256 + hf * 16;
  for (int kk = 0; kk < 256; kk += 32) {
    Frag af;
    af.u[0] = *(const u32x4*)(ap + kk);
    af.u[1] = *(const u32x4*)(ap + kk + 8);
#pragma unroll
    for (int t = 0; t < 8; t++) {
      Frag bf;
      const u16* bp = Bb + (size_t)(j0 + t * 16 + l15) * 256 + kk + hf * 16;
      bf.u[0] = *(const u32x4*)(bp);
      bf.u[1] = *(const u32x4*)(bp + 8);
      acc[t] = WMMA_BF16(af.v, bf.v, acc[t]);
    }
  }
#pragma unroll
  for (int t = 0; t < 8; t++) {
    int j = j0 + t * 16 + l15;
#pragma unroll
    for (int r = 0; r < 8; r++) {
      int i = i0 + r + hf * 8;
      pm[((size_t)i * 256 + j) * 128 + c] = acc[t][r];
    }
  }
}

// ---------------- Fused flash attention over one (i, h) block ---------------
// Wave = 16 q-rows; k loop in chunks of 32; online softmax; P via LDS.
__global__ __launch_bounds__(256) void attn_k(const u16* __restrict__ Qb,
                                              const u16* __restrict__ Kb,
                                              const u16* __restrict__ Vt,
                                              const float* __restrict__ tb,
                                              const float* __restrict__ mask,
                                              float* __restrict__ O, int tflag) {
  __shared__ __attribute__((aligned(16))) u16 plds[8][512];
  int wv = threadIdx.x >> 5, lane = threadIdx.x & 31;
  int hf = lane >> 4, l15 = lane & 15;
  int i = blockIdx.z, h = blockIdx.y;
  int q0 = blockIdx.x * 128 + wv * 16;

  Frag qf;
  {
    const u16* qp = Qb + (size_t)(i * 256 + q0 + l15) * 128 + h * 32 + hf * 16;
    qf.u[0] = *(const u32x4*)qp;
    qf.u[1] = *(const u32x4*)(qp + 8);
  }

  float mr[8], lr[8], al[8];
#pragma unroll
  for (int r = 0; r < 8; r++) { mr[r] = -3e38f; lr[r] = 0.f; }
  v8f o0 = (v8f){0, 0, 0, 0, 0, 0, 0, 0};
  v8f o1 = (v8f){0, 0, 0, 0, 0, 0, 0, 0};
  const float scale = 0.17677669529663687f;  // 1/sqrt(32)
  const v8f zero = (v8f){0, 0, 0, 0, 0, 0, 0, 0};

  for (int kt = 0; kt < 8; kt++) {
    v8f s0, s1;
    {
      Frag kf, kf2;
      const u16* kp  = Kb + (size_t)(i * 256 + kt * 32 + l15) * 128 + h * 32 + hf * 16;
      const u16* kp2 = Kb + (size_t)(i * 256 + kt * 32 + 16 + l15) * 128 + h * 32 + hf * 16;
      kf.u[0]  = *(const u32x4*)kp;   kf.u[1]  = *(const u32x4*)(kp + 8);
      kf2.u[0] = *(const u32x4*)kp2;  kf2.u[1] = *(const u32x4*)(kp2 + 8);
      s0 = WMMA_BF16(qf.v, kf.v, zero);
      s1 = WMMA_BF16(qf.v, kf2.v, zero);
    }
#pragma unroll
    for (int r = 0; r < 8; r++) {
      int qr = q0 + r + hf * 8;
      int c0 = kt * 32 + l15, c1 = c0 + 16;
      float mv0 = tflag ? mask[c0 * 256 + i] : mask[i * 256 + c0];
      float mv1 = tflag ? mask[c1 * 256 + i] : mask[i * 256 + c1];
      s0[r] = s0[r] * scale + tb[h * 65536 + qr * 256 + c0] + 1e9f * (mv0 - 1.f);
      s1[r] = s1[r] * scale + tb[h * 65536 + qr * 256 + c1] + 1e9f * (mv1 - 1.f);
      float cm = hmax16(fmaxf(s0[r], s1[r]));
      float mn = fmaxf(mr[r], cm);
      al[r] = __expf(mr[r] - mn);
      mr[r] = mn;
      float p0 = __expf(s0[r] - mn), p1 = __expf(s1[r] - mn);
      float rs = hsum16(p0 + p1);
      lr[r] = lr[r] * al[r] + rs;
      o0[r] *= al[r];
      o1[r] *= al[r];
      plds[wv][(r + hf * 8) * 32 + l15]      = f2bf(p0);
      plds[wv][(r + hf * 8) * 32 + 16 + l15] = f2bf(p1);
    }
    asm volatile("s_wait_dscnt 0" ::: "memory");  // wave-private LDS round trip
    Frag pf;
    pf.u[0] = *(const u32x4*)&plds[wv][l15 * 32 + hf * 16];
    pf.u[1] = *(const u32x4*)&plds[wv][l15 * 32 + hf * 16 + 8];
    Frag vf0, vf1;
    {
      const u16* vp  = Vt + (size_t)((i * 4 + h) * 32 + l15) * 256 + kt * 32 + hf * 16;
      const u16* vp1 = Vt + (size_t)((i * 4 + h) * 32 + 16 + l15) * 256 + kt * 32 + hf * 16;
      vf0.u[0] = *(const u32x4*)vp;   vf0.u[1] = *(const u32x4*)(vp + 8);
      vf1.u[0] = *(const u32x4*)vp1;  vf1.u[1] = *(const u32x4*)(vp1 + 8);
    }
    o0 = WMMA_BF16(pf.v, vf0.v, o0);
    o1 = WMMA_BF16(pf.v, vf1.v, o1);
  }

#pragma unroll
  for (int r = 0; r < 8; r++) {
    float inv = 1.f / lr[r];
    size_t base = (size_t)(i * 256 + q0 + r + hf * 8) * 128 + h * 32;
    O[base + l15]      = o0[r] * inv;
    O[base + 16 + l15] = o1[r] * inv;
  }
}

// ============================ host orchestration ===========================
extern "C" void kernel_launch(void* const* d_in, const int* in_sizes, int n_in,
                              void* d_out, int out_size, void* d_ws, size_t ws_size,
                              hipStream_t stream) {
  (void)in_sizes; (void)n_in; (void)out_size; (void)ws_size;
  const int M = 65536;
  auto F = [&](int i) { return (const float*)d_in[i]; };
  const float* pair = F(0);
  const float* mask = F(1);

  char* ws = (char*)d_ws;
  u16*   Zbf = (u16*)(ws + 0);                      // 16 MB bf16 [M,128]
  float* T0  = (float*)(ws + (16ull << 20));        // 32 MB f32  [M,128]
  float* T3  = (float*)(ws + (48ull << 20));        // 32 MB f32  [M,128] (gates)
  u16*   T1  = (u16*)(ws + (80ull << 20));          // 64 MB bf16 (left / Q,K,V / hidden)
  u16*   T2  = (u16*)(ws + (144ull << 20));         // 16 MB bf16 (right / g*o)
  float* PA  = (float*)(ws + (160ull << 20));       // 32 MB f32 pair ping
  float* TB  = (float*)(ws + (192ull << 20));       // 1 MB  f32 tri_bias [4][M]
  u16*   WS0 = (u16*)(ws + (193ull << 20));         // transposed bf16 weights
  auto wslot = [&](int i) { return WS0 + (size_t)i * 65536; };

  float* PB = (float*)d_out;
  u16* Qb  = T1;
  u16* Kb  = T1 + (size_t)M * 128;
  u16* Vt  = T1 + (size_t)2 * M * 128;
  u16* GOb = T2;
  u16* Hb  = T1;  // [M,512] bf16 hidden for transition

  dim3 blk(256);
  auto prep = [&](const float* W, u16* Wt, int K, int N) {
    prep_wt<<<(K * N + 255) / 256, blk, 0, stream>>>(W, Wt, K, N);
  };

  // ---- prep all transposed bf16 weights (24 slots) ----
  int mb[2] = {2, 18};   // tmo, tmi bases
  for (int s = 0; s < 2; s++) {
    int pb = mb[s], o = s * 6;
    prep(F(pb + 2),  wslot(o + 0), 128, 128);  // Wl
    prep(F(pb + 6),  wslot(o + 1), 128, 128);  // Wlg
    prep(F(pb + 4),  wslot(o + 2), 128, 128);  // Wr
    prep(F(pb + 8),  wslot(o + 3), 128, 128);  // Wrg
    prep(F(pb + 10), wslot(o + 4), 128, 128);  // Wg
    prep(F(pb + 14), wslot(o + 5), 128, 128);  // Wo
  }
  int ab[2] = {34, 44};  // tas, tae bases
  for (int s = 0; s < 2; s++) {
    int pb = ab[s], o = 12 + s * 5;
    prep(F(pb + 3), wslot(o + 0), 128, 128);   // Wq
    prep(F(pb + 4), wslot(o + 1), 128, 128);   // Wk
    prep(F(pb + 5), wslot(o + 2), 128, 128);   // Wv
    prep(F(pb + 6), wslot(o + 3), 128, 128);   // Wg
    prep(F(pb + 8), wslot(o + 4), 128, 128);   // Wout
  }
  prep(F(56), wslot(22), 128, 512);  // W1t [512][128]
  prep(F(58), wslot(23), 512, 128);  // W2t [128][512]

  dim3 g1(512, 1), g4(512, 4), ge(2, 2, 128), ga(2, 4, 256);

  // ---- triangle multiplication stage ----
  auto tri_mult = [&](const float* Pin, float* Pout, int pb, int wsl, bool outgoing) {
    int tf = outgoing ? 0 : 1;
    ln_k<<<8192, blk, 0, stream>>>(Pin, F(pb + 0), F(pb + 1), Zbf, 0);
    gemm_k<0><<<g1, blk, 0, stream>>>(Zbf, 128, wslot(wsl + 1), F(pb + 7), nullptr, nullptr, nullptr, T3, nullptr, 128, 0);
    gemm_k<1><<<g1, blk, 0, stream>>>(Zbf, 128, wslot(wsl + 0), F(pb + 3), mask, T3, nullptr, nullptr, T1, 128, tf);
    gemm_k<0><<<g1, blk, 0, stream>>>(Zbf, 128, wslot(wsl + 3), F(pb + 9), nullptr, nullptr, nullptr, T3, nullptr, 128, 0);
    gemm_k<1><<<g1, blk, 0, stream>>>(Zbf, 128, wslot(wsl + 2), F(pb + 5), mask, T3, nullptr, nullptr, T2, 128, tf);
    gemm_k<0><<<g1, blk, 0, stream>>>(Zbf, 128, wslot(wsl + 4), F(pb + 11), nullptr, nullptr, nullptr, T3, nullptr, 128, 0);
    einsum_k<<<ge, blk, 0, stream>>>(T1, T2, T0);
    ln_k<<<8192, blk, 0, stream>>>(T0, F(pb + 12), F(pb + 13), Zbf, 0);
    gemm_k<4><<<g1, blk, 0, stream>>>(Zbf, 128, wslot(wsl + 5), F(pb + 15), nullptr, T3, Pin, Pout, nullptr, 128, 0);
  };

  // ---- triangle attention stage ----
  auto tri_attn = [&](const float* Pin, float* Pout, int pb, int wsl, int tf) {
    ln_k<<<8192, blk, 0, stream>>>(Pin, F(pb + 0), F(pb + 1), Zbf, tf);
    tb_k<<<8192, blk, 0, stream>>>(Zbf, F(pb + 2), TB);
    gemm_k<5><<<g1, blk, 0, stream>>>(Zbf, 128, wslot(wsl + 0), nullptr, nullptr, nullptr, nullptr, nullptr, Qb, 128, 0);
    gemm_k<5><<<g1, blk, 0, stream>>>(Zbf, 128, wslot(wsl + 1), nullptr, nullptr, nullptr, nullptr, nullptr, Kb, 128, 0);
    gemm_k<6><<<g1, blk, 0, stream>>>(Zbf, 128, wslot(wsl + 2), nullptr, nullptr, nullptr, nullptr, nullptr, Vt, 128, 0);
    gemm_k<0><<<g1, blk, 0, stream>>>(Zbf, 128, wslot(wsl + 3), F(pb + 7), nullptr, nullptr, nullptr, T3, nullptr, 128, 0);
    attn_k<<<ga, blk, 0, stream>>>(Qb, Kb, Vt, TB, mask, T0, tf);
    gatemul_k<<<8192, blk, 0, stream>>>(T3, T0, GOb);
    gemm_k<3><<<g1, blk, 0, stream>>>(GOb, 128, wslot(wsl + 4), F(pb + 9), nullptr, nullptr, Pin, Pout, nullptr, 128, tf);
  };

  // ---- pipeline: mult(out) -> mult(in) -> attn(row) -> attn(col) -> transition
  tri_mult(pair, PA, 2, 0, true);
  tri_mult(PA, PB, 18, 6, false);
  tri_attn(PB, PA, 34, 12, 0);
  tri_attn(PA, PB, 44, 17, 1);

  // transition (in-place on PB = d_out)
  ln_k<<<8192, blk, 0, stream>>>(PB, F(54), F(55), Zbf, 0);
  gemm_k<2><<<g4, blk, 0, stream>>>(Zbf, 128, wslot(22), F(57), nullptr, nullptr, nullptr, nullptr, Hb, 512, 0);
  gemm_k<3><<<g1, blk, 0, stream>>>(Hb, 512, wslot(23), F(59), nullptr, nullptr, PB, PB, nullptr, 128, 0);
}
